// GRUResidual_30210799960396
// MI455X (gfx1250) — matrix operand order repaired and checked by reading, not compiled, not run on hardware
//
#include <hip/hip_runtime.h>
#include <hip/hip_bf16.h>

// Persistent-wave GRU trajectory kernel for gfx1250 (MI455X).
// One wave = 16 batch rows, full 2048-step recurrence on-chip.
// Recurrent matmul: v_wmma_f32_16x16x32_f16 (W_hh in LDS as f16).
// Input path + biases: v_wmma_f32_16x16x4_f32 with A=[z,1], B=[W_ih|bias].
// Head: 2x v_wmma_f32_16x16x32_f16 with zero-padded 3-col B in registers.

#define ROWS  16     // batch rows per wave (WMMA M)
#define WAVES 4      // waves per block
#define HID   64
#define GATES 192    // 3*HID
#define WST   72     // W_hh LDS row stride in f16 elems (padded: bank-friendly)
#define HST   72     // h staging row stride in f16 elems

typedef __attribute__((ext_vector_type(16))) _Float16 v16h;
typedef __attribute__((ext_vector_type(8)))  _Float16 v8h;
typedef __attribute__((ext_vector_type(8)))  float    v8f;
typedef __attribute__((ext_vector_type(2)))  float    v2f;

#if __has_builtin(__builtin_amdgcn_wmma_f32_16x16x4_f32)
#define HAS_F32X4 1
#else
#define HAS_F32X4 0
#endif

__device__ __forceinline__ v16h cat8(v8h a, v8h b) {
  return __builtin_shufflevector(a, b, 0,1,2,3,4,5,6,7,8,9,10,11,12,13,14,15);
}

__device__ __forceinline__ v8f wmma16(v16h a, v16h b, v8f c) {
  // D = A(16x32 f16) * B(32x16 f16) + C(16x16 f32)
  return __builtin_amdgcn_wmma_f32_16x16x32_f16(false, a, false, b,
                                                (short)0, c, false, false);
}

#if HAS_F32X4
__device__ __forceinline__ v8f wmma4(v2f a, v2f b, v8f c) {
  // D = A(16x4 f32) * B(4x16 f32) + C(16x16 f32)
  return __builtin_amdgcn_wmma_f32_16x16x4_f32(false, a, false, b,
                                               (short)0, c, false, false);
}
#endif

__device__ __forceinline__ float tanh_fast(float x) {
#if __has_builtin(__builtin_amdgcn_tanhf)
  return __builtin_amdgcn_tanhf(x);          // gfx1250 v_tanh_f32
#elif __has_builtin(__builtin_amdgcn_tanh_f32)
  return __builtin_amdgcn_tanh_f32(x);
#else
  float e = __expf(x + x);
  return 1.0f - 2.0f / (e + 1.0f);
#endif
}

__device__ __forceinline__ float sigm(float x) {
  return fmaf(0.5f, tanh_fast(0.5f * x), 0.5f);
}

// B operand tile for 16x16x32 f16 WMMA: lane n = lane%16 holds 16 contiguous
// K values (K-half selected by lane/16) -> a row of row-major W_hh in LDS.
__device__ __forceinline__ v16h loadB16(const _Float16* p) {
  return cat8(*(const v8h*)p, *(const v8h*)(p + 8));
}

// A operand: lane m = lane%16, K-chunk q (32 wide):
// 16 halves = K in [32q+16*hi, +16).
__device__ __forceinline__ v16h loadA16(const _Float16* row, int q, int hi) {
  const _Float16* p = row + 32 * q + 16 * hi;
  return cat8(*(const v8h*)p, *(const v8h*)(p + 8));
}

__global__ __launch_bounds__(WAVES * 32) void gru_traj_kernel(
    const float* __restrict__ z0g, const float* __restrict__ dtp,
    const int* __restrict__ stepsp,
    const float* __restrict__ W_ih, const float* __restrict__ W_hh,
    const float* __restrict__ b_ih, const float* __restrict__ b_hh,
    const float* __restrict__ W_head, const float* __restrict__ b_head,
    float* __restrict__ traj)
{
  __shared__ _Float16 whh_s[GATES * WST];          // W_hh row-major, f16
  __shared__ _Float16 hst_s[WAVES][ROWS * HST];    // per-wave h staging (f16)
  __shared__ float    fst_s[WAVES][ROWS * 4];      // per-wave head staging
#if !HAS_F32X4
  __shared__ float    zst_s[WAVES][ROWS * 4];      // per-wave z staging
#endif

  const int tid  = threadIdx.x;
  const int w    = tid >> 5;         // wave in block (warpSize == 32)
  const int lane = tid & 31;
  const int ln   = lane & 15;
  const int hi   = lane >> 4;

  const float dtv  = dtp[0];
  const int steps  = stepsp[0];
  const int S1     = steps + 1;

  // ---- prologue: convert W_hh -> f16 in LDS (row-major, padded stride) ----
  for (int i = tid; i < GATES * HID; i += WAVES * 32) {
    int c = i >> 6, k = i & 63;
    whh_s[c * WST + k] = (_Float16)W_hh[i];
  }

  // ---- per-lane constant registers ----
#if HAS_F32X4
  // zb[jt]: B(4x16) tile for the f32 K=4 WMMA. Rows: Wih[:,0],Wih[:,1],Wih[:,2],bias.
  // r/u tiles (jt<8): bias = b_ih+b_hh (fused). n tiles (jt>=8): bias = b_ih only
  // (b_hh_n is added inside the r* term per PyTorch GRU).
  v2f zb[12];
#pragma unroll
  for (int jt = 0; jt < 12; ++jt) {
    int c = 16 * jt + ln;
    float w0 = W_ih[c * 3 + 0], w1 = W_ih[c * 3 + 1], w2 = W_ih[c * 3 + 2];
    float bias = (jt < 8) ? (b_ih[c] + b_hh[c]) : b_ih[c];
    zb[jt].x = hi ? w2   : w0;
    zb[jt].y = hi ? bias : w1;
  }
#else
  float wiw[12][3], wib[12];
#pragma unroll
  for (int jt = 0; jt < 12; ++jt) {
    int c = 16 * jt + ln;
    wiw[jt][0] = W_ih[c * 3 + 0];
    wiw[jt][1] = W_ih[c * 3 + 1];
    wiw[jt][2] = W_ih[c * 3 + 2];
    wib[jt] = (jt < 8) ? (b_ih[c] + b_hh[c]) : b_ih[c];
  }
#endif
  float bhhn[4];
#pragma unroll
  for (int j = 0; j < 4; ++j) bhhn[j] = b_hh[128 + 16 * j + ln];

  // Head B operand (W_head^T, 3 valid cols zero-padded to 16), in registers.
  v16h whB0 = {}, whB1 = {};
  if (ln < 3) {
#pragma unroll
    for (int i = 0; i < 16; ++i) {
      whB0[i] = (_Float16)W_head[ln * 64 + 16 * hi + i];
      whB1[i] = (_Float16)W_head[ln * 64 + 32 + 16 * hi + i];
    }
  }
  const float bh0 = b_head[0], bh1 = b_head[1], bh2 = b_head[2];

  // ---- per-lane state: z for row m = ln (duplicated across halves) ----
  const int gbase = (blockIdx.x * WAVES + w) * ROWS;
  const int brow  = gbase + ln;
  float z0r = z0g[brow * 3 + 0];
  float z1r = z0g[brow * 3 + 1];
  float z2r = z0g[brow * 3 + 2];

  if (hi == 0) {  // traj[:, 0, :] = z0
    size_t o = (size_t)brow * S1 * 3;
    traj[o] = z0r; traj[o + 1] = z1r; traj[o + 2] = z2r;
  }
#if !HAS_F32X4
  if (hi == 0) {
    zst_s[w][ln * 4 + 0] = z0r;
    zst_s[w][ln * 4 + 1] = z1r;
    zst_s[w][ln * 4 + 2] = z2r;
  }
#endif
  __syncthreads();

  // h state: C-layout f32 (lane holds (m = e+8*hi, n = 16j+ln)), starts at 0.
  v8f hreg[4] = {v8f{}, v8f{}, v8f{}, v8f{}};
  v16h A0 = {};  // h in A-layout f16, K 0..31  (h0 == 0)
  v16h A1 = {};  // K 32..63

  const _Float16* myArow = &hst_s[w][ln * HST];

  for (int t = 0; t < steps; ++t) {
    v2f az;
    az.x = hi ? z2r : z0r;
    az.y = hi ? 1.0f : z1r;

#pragma unroll
    for (int j = 0; j < 4; ++j) {
      // gate pre-activations for 16 hidden cols (16j..16j+15)
      const _Float16* pR = &whh_s[(      16 * j + ln) * WST + 16 * hi];
      const _Float16* pU = &whh_s[( 64 + 16 * j + ln) * WST + 16 * hi];
      const _Float16* pN = &whh_s[(128 + 16 * j + ln) * WST + 16 * hi];

      v8f aR = {}, aU = {}, aNh = {};
      aR  = wmma16(A0, loadB16(pR), aR);
      aR  = wmma16(A1, loadB16(pR + 32), aR);
      aU  = wmma16(A0, loadB16(pU), aU);
      aU  = wmma16(A1, loadB16(pU + 32), aU);
      aNh = wmma16(A0, loadB16(pN), aNh);
      aNh = wmma16(A1, loadB16(pN + 32), aNh);

#if HAS_F32X4
      aR = wmma4(az, zb[j], aR);            // += z@Wih_r^T + (b_ih+b_hh)_r
      aU = wmma4(az, zb[j + 4], aU);        // += z@Wih_z^T + (b_ih+b_hh)_z
      v8f aNi = {};
      aNi = wmma4(az, zb[j + 8], aNi);      //  = z@Wih_n^T + b_ih_n
#else
      float aNi_f[8];
#pragma unroll
      for (int e = 0; e < 8; ++e) {
        int m = e + 8 * hi;
        float zm0 = zst_s[w][m * 4 + 0];
        float zm1 = zst_s[w][m * 4 + 1];
        float zm2 = zst_s[w][m * 4 + 2];
        aR[e] += fmaf(zm0, wiw[j][0], fmaf(zm1, wiw[j][1],
                      fmaf(zm2, wiw[j][2], wib[j])));
        aU[e] += fmaf(zm0, wiw[j + 4][0], fmaf(zm1, wiw[j + 4][1],
                      fmaf(zm2, wiw[j + 4][2], wib[j + 4])));
        aNi_f[e] = fmaf(zm0, wiw[j + 8][0], fmaf(zm1, wiw[j + 8][1],
                        fmaf(zm2, wiw[j + 8][2], wib[j + 8])));
      }
#endif

#pragma unroll
      for (int e = 0; e < 8; ++e) {
#if HAS_F32X4
        float ni = aNi[e];
#else
        float ni = aNi_f[e];
#endif
        float r = sigm(aR[e]);
        float u = sigm(aU[e]);
        float n = tanh_fast(fmaf(r, aNh[e] + bhhn[j], ni));
        float hn = fmaf(u, hreg[j][e] - n, n);    // (1-u)*n + u*h
        hreg[j][e] = hn;
        hst_s[w][(e + 8 * hi) * HST + 16 * j + ln] = (_Float16)hn;
      }
    }

    // wave-private LDS: order staging stores before cross-lane A reload
    asm volatile("s_wait_dscnt 0x0" ::: "memory");
    A0 = loadA16(myArow, 0, hi);
    A1 = loadA16(myArow, 1, hi);

    // head: f_hat = h_new @ W_head^T  (2 WMMAs, 3 valid cols)
    v8f fh = {};
    fh = wmma16(A0, whB0, fh);
    fh = wmma16(A1, whB1, fh);
    if (ln < 3) {
#pragma unroll
      for (int e = 0; e < 8; ++e)
        fst_s[w][(e + 8 * hi) * 4 + ln] = fh[e];
    }
    asm volatile("s_wait_dscnt 0x0" ::: "memory");

    float f0 = fst_s[w][ln * 4 + 0] + bh0;
    float f1 = fst_s[w][ln * 4 + 1] + bh1;
    float f2 = fst_s[w][ln * 4 + 2] + bh2;
    z0r = fmaf(dtv, f0, z0r);
    z1r = fmaf(dtv, f1, z1r);
    z2r = fmaf(dtv, f2, z2r);

#if !HAS_F32X4
    if (hi == 0) {
      zst_s[w][ln * 4 + 0] = z0r;
      zst_s[w][ln * 4 + 1] = z1r;
      zst_s[w][ln * 4 + 2] = z2r;
    }
    asm volatile("s_wait_dscnt 0x0" ::: "memory");
#endif

    if (hi == 0) {
      size_t o = ((size_t)brow * S1 + (t + 1)) * 3;
      traj[o] = z0r; traj[o + 1] = z1r; traj[o + 2] = z2r;
    }
  }
}

extern "C" void kernel_launch(void* const* d_in, const int* in_sizes, int n_in,
                              void* d_out, int out_size, void* d_ws, size_t ws_size,
                              hipStream_t stream) {
  const float* z0     = (const float*)d_in[0];
  const float* dtp    = (const float*)d_in[1];
  const int*   stepsp = (const int*)d_in[2];
  const float* W_ih   = (const float*)d_in[3];
  const float* W_hh   = (const float*)d_in[4];
  const float* b_ih   = (const float*)d_in[5];
  const float* b_hh   = (const float*)d_in[6];
  const float* W_head = (const float*)d_in[7];
  const float* b_head = (const float*)d_in[8];
  float* traj = (float*)d_out;

  int B = in_sizes[0] / 3;               // z0 is [B,3]
  int blocks = B / (ROWS * WAVES);       // B=16384 -> 256 blocks of 128 threads
  hipLaunchKernelGGL(gru_traj_kernel, dim3(blocks), dim3(WAVES * 32), 0, stream,
                     z0, dtp, stepsp, W_ih, W_hh, b_ih, b_hh, W_head, b_head,
                     traj);
}
